// ConvTemporalGraphical_1262720385245
// MI455X (gfx1250) — matrix operand order repaired and checked by reading, not compiled, not run on hardware
//
#include <hip/hip_runtime.h>

typedef __attribute__((ext_vector_type(16))) __bf16 v16bf;
typedef __attribute__((ext_vector_type(8)))  __bf16 v8bf;
typedef __attribute__((ext_vector_type(8)))  float  v8f;

#define NB   32
#define CINC 64
#define COUTC 64
#define TT   6
#define VV   512
#define TV   (TT * VV)          // 3072
#define ROWS (COUTC * TT)       // 384 (ct rows of xc/out)

// ---------------------------------------------------------------------------
// k1: xc[n, o, q] = sum_i conv_w[o,i] * x[n, i, q] + conv_b[o]   (q = t*V+v)
// One 16x16 tile per wave, K=64 -> 2x v_wmma_f32_16x16x32_bf16.
// Stored as bf16 in (n, 384, 512) row-major (== (n, o, t, v)).
// ---------------------------------------------------------------------------
__global__ __launch_bounds__(256) void k1_conv_gemm(
    const float* __restrict__ x,       // (N, CIN, TV)
    const float* __restrict__ conv_w,  // (COUT, CIN)
    const float* __restrict__ conv_b,  // (COUT)
    __bf16* __restrict__ xc)           // (N, COUT, TV) bf16
{
    const int lane = threadIdx.x & 31;
    const int wave = threadIdx.x >> 5;
    const int tile = blockIdx.x * 8 + wave;     // 32 * 4 * 192 = 24576 tiles
    const int qt  = tile % 192;
    const int rem = tile / 192;
    const int mt  = rem % 4;
    const int n   = rem / 4;

    const int half = lane >> 4;
    const int r    = lane & 15;
    const int m0   = mt * 16;
    const int q0   = qt * 16;

    v8f c;
    #pragma unroll
    for (int v = 0; v < 8; ++v) c[v] = conv_b[m0 + v + 8 * half];

    const float* xn = x + (size_t)n * CINC * TV;

    #pragma unroll
    for (int kb = 0; kb < CINC; kb += 32) {
        v16bf a, b;
        // A fragment: row o = m0+r; elems 0..7 -> i = kb+half*8+e, 8..15 -> i = kb+16+half*8+e
        const float* wrow = conv_w + (size_t)(m0 + r) * CINC + kb;
        #pragma unroll
        for (int e = 0; e < 8; ++e) {
            a[e]     = (__bf16)wrow[half * 8 + e];
            a[8 + e] = (__bf16)wrow[16 + half * 8 + e];
        }
        // B fragment: col q = q0+r; elems 0..7 -> i = kb+half*8+e, 8..15 -> i = kb+16+half*8+(e-8)
        #pragma unroll
        for (int e = 0; e < 16; ++e)
            b[e] = (__bf16)xn[(size_t)(kb + half * 8 + e + (e & 8)) * TV + q0 + r];

        c = __builtin_amdgcn_wmma_f32_16x16x32_bf16(false, a, false, b,
                                                    (short)0, c, false, false);
    }

    __bf16* xcn = xc + (size_t)n * COUTC * TV;
    #pragma unroll
    for (int v = 0; v < 8; ++v)
        xcn[(size_t)(m0 + v + 8 * half) * TV + q0 + r] = (__bf16)c[v];
}

// ---------------------------------------------------------------------------
// k2: per (n,v): x10[c] = sum_t xc[n,c,t,v]*l2_w[t] + l2_b ;
//     s_i = dot(x10, l1_w[0:64]) ; s_j = dot(x10, l1_w[64:128])
// ---------------------------------------------------------------------------
__global__ __launch_bounds__(256) void k2_scores(
    const __bf16* __restrict__ xc,     // (N, COUT, TV)
    const float* __restrict__ l2_w,    // (T)
    const float* __restrict__ l2_b,    // (1)
    const float* __restrict__ l1_w,    // (2*COUT)
    float* __restrict__ s_i,           // (N*V)
    float* __restrict__ s_j)           // (N*V)
{
    const int idx = blockIdx.x * blockDim.x + threadIdx.x;   // n*512 + v
    const int v = idx & (VV - 1);
    const int n = idx >> 9;

    float w2[TT];
    #pragma unroll
    for (int t = 0; t < TT; ++t) w2[t] = l2_w[t];
    const float b2 = l2_b[0];

    float si = 0.f, sj = 0.f;
    const __bf16* base = xc + (size_t)n * COUTC * TV + v;
    for (int cc = 0; cc < COUTC; ++cc) {
        float acc = 0.f;
        #pragma unroll
        for (int t = 0; t < TT; ++t)
            acc += (float)base[(size_t)cc * TV + t * VV] * w2[t];
        acc += b2;
        si += acc * l1_w[cc];
        sj += acc * l1_w[COUTC + cc];
    }
    s_i[idx] = si;
    s_j[idx] = sj;
}

// ---------------------------------------------------------------------------
// k3: masked leaky-relu softmax row (n,i) over j=0..511.
// Writes a4 f32 straight into the d_out tail (coalesced).
// ---------------------------------------------------------------------------
__global__ __launch_bounds__(256) void k3_softmax(
    const float* __restrict__ A,       // (N, 8, V, V); channel 7 = mask
    const float* __restrict__ s_i,
    const float* __restrict__ s_j,
    const float* __restrict__ l1_b,    // (1)
    float* __restrict__ a4_out)        // (N, V, V) f32 -> d_out tail
{
    __shared__ float red[256];
    const int row = blockIdx.x;            // n*512 + i
    const int n = row >> 9;
    const int tid = threadIdx.x;

    const float sii = s_i[row];
    const float b1  = l1_b[0];
    const float* mrow = A + (((size_t)n * 8 + 7) * VV + (row & 511)) * VV;
    const float* sjn  = s_j + (size_t)n * VV;

    float vals[2];
    #pragma unroll
    for (int u = 0; u < 2; ++u) {
        const int j = tid + u * 256;
        float s = sii + sjn[j] + b1;
        s = (s > 0.f) ? s : 0.2f * s;                 // leaky relu 0.2
        if ((int)mrow[j] == 0) s = -1000.0f;          // NEG_FILL
        vals[u] = s;
    }

    // row max
    red[tid] = fmaxf(vals[0], vals[1]);
    __syncthreads();
    for (int off = 128; off > 0; off >>= 1) {
        if (tid < off) red[tid] = fmaxf(red[tid], red[tid + off]);
        __syncthreads();
    }
    const float rmax = red[0];
    __syncthreads();

    const float e0 = __expf(vals[0] - rmax);
    const float e1 = __expf(vals[1] - rmax);
    red[tid] = e0 + e1;
    __syncthreads();
    for (int off = 128; off > 0; off >>= 1) {
        if (tid < off) red[tid] += red[tid + off];
        __syncthreads();
    }
    const float inv = 1.0f / red[0];

    float* orow = a4_out + (size_t)row * VV;
    orow[tid]       = e0 * inv;
    orow[tid + 256] = e1 * inv;
}

// ---------------------------------------------------------------------------
// k3b: LDS-tiled transpose + bf16 convert: a4T[n, w, v] = (bf16) a4[n, v, w]
// Gives the k4 B-matrix contiguous-K rows (one 32B vector load / fragment).
// block (32,8), grid (16,16,N)
// ---------------------------------------------------------------------------
__global__ __launch_bounds__(256) void k3b_transpose(
    const float* __restrict__ a4,      // (N, V, V) f32
    __bf16* __restrict__ a4T)          // (N, V, V) bf16, [n][w][v]
{
    __shared__ float tile[32][33];
    const int n  = blockIdx.z;
    const int i0 = blockIdx.x * 32;    // source row (v) block
    const int j0 = blockIdx.y * 32;    // source col (w) block

    const float* src = a4 + ((size_t)n * VV + i0) * VV + j0;
    #pragma unroll
    for (int s = 0; s < 32; s += 8)
        tile[threadIdx.y + s][threadIdx.x] =
            src[(size_t)(threadIdx.y + s) * VV + threadIdx.x];
    __syncthreads();

    __bf16* dst = a4T + ((size_t)n * VV + j0) * VV + i0;
    #pragma unroll
    for (int s = 0; s < 32; s += 8)
        dst[(size_t)(threadIdx.y + s) * VV + threadIdx.x] =
            (__bf16)tile[threadIdx.x][threadIdx.y + s];
}

// ---------------------------------------------------------------------------
// k4: out[n, ct, w] = sum_v xc[n, ct, v] * a4[n, v, w]
// 2x2 register-blocked: one wave computes a 32x32 output block.
// K=512 -> 16 slabs x 4 WMMA; A and B fragments are pure b128 vector loads.
// ---------------------------------------------------------------------------
__global__ __launch_bounds__(256) void k4_out_gemm(
    const __bf16* __restrict__ xc,     // (N, 384, 512)  A, row-major
    const __bf16* __restrict__ a4T,    // (N, 512, 512)  B^T: [n][w][v]
    float* __restrict__ out)           // (N, 384, 512) f32
{
    const int lane = threadIdx.x & 31;
    const int wave = threadIdx.x >> 5;
    const int tile = blockIdx.x * 8 + wave;     // 32 * 12 * 16 = 6144 tiles
    const int wt  = tile % 16;
    const int rem = tile / 16;
    const int mt  = rem % 12;
    const int n   = rem / 12;

    const int half = lane >> 4;
    const int r    = lane & 15;
    const int m0   = mt * 32;
    const int w0   = wt * 32;

    v8f c00 = {}, c01 = {}, c10 = {}, c11 = {};

    const __bf16* arow0 = xc  + ((size_t)n * ROWS + m0 + r) * VV;
    const __bf16* arow1 = arow0 + (size_t)16 * VV;
    const __bf16* brow0 = a4T + ((size_t)n * VV + w0 + r) * VV;
    const __bf16* brow1 = brow0 + (size_t)16 * VV;

    #pragma unroll 2
    for (int kb = 0; kb < VV; kb += 32) {
        __builtin_prefetch(arow0 + kb + 128, 0, 0);
        __builtin_prefetch(brow0 + kb + 128, 0, 0);

        v16bf a0, a1, b0, b1;
        // A fragments: elems 0..15 -> contiguous K = kb+half*16+e
        const v8bf a0lo = *(const v8bf*)(arow0 + kb + half * 16);
        const v8bf a0hi = *(const v8bf*)(arow0 + kb + half * 16 + 8);
        const v8bf a1lo = *(const v8bf*)(arow1 + kb + half * 16);
        const v8bf a1hi = *(const v8bf*)(arow1 + kb + half * 16 + 8);
        #pragma unroll
        for (int e = 0; e < 8; ++e) {
            a0[e] = a0lo[e]; a0[8 + e] = a0hi[e];
            a1[e] = a1lo[e]; a1[8 + e] = a1hi[e];
        }
        // B fragments: lane = col w0(+16)+r, elems = contiguous K kb+half*16+e
        b0 = *(const v16bf*)(brow0 + kb + half * 16);
        b1 = *(const v16bf*)(brow1 + kb + half * 16);

        c00 = __builtin_amdgcn_wmma_f32_16x16x32_bf16(false, a0, false, b0,
                                                      (short)0, c00, false, false);
        c01 = __builtin_amdgcn_wmma_f32_16x16x32_bf16(false, a0, false, b1,
                                                      (short)0, c01, false, false);
        c10 = __builtin_amdgcn_wmma_f32_16x16x32_bf16(false, a1, false, b0,
                                                      (short)0, c10, false, false);
        c11 = __builtin_amdgcn_wmma_f32_16x16x32_bf16(false, a1, false, b1,
                                                      (short)0, c11, false, false);
    }

    float* obase = out + (size_t)n * ROWS * VV;
    #pragma unroll
    for (int v = 0; v < 8; ++v) {
        const size_t row_lo = (size_t)(m0 + v + 8 * half) * VV;
        const size_t row_hi = (size_t)(m0 + 16 + v + 8 * half) * VV;
        obase[row_lo + w0 + r]       = c00[v];
        obase[row_lo + w0 + 16 + r]  = c01[v];
        obase[row_hi + w0 + r]       = c10[v];
        obase[row_hi + w0 + 16 + r]  = c11[v];
    }
}

// ---------------------------------------------------------------------------
extern "C" void kernel_launch(void* const* d_in, const int* in_sizes, int n_in,
                              void* d_out, int out_size, void* d_ws, size_t ws_size,
                              hipStream_t stream) {
    (void)in_sizes; (void)n_in; (void)out_size; (void)ws_size;

    const float* x      = (const float*)d_in[0];
    const float* A      = (const float*)d_in[1];
    const float* conv_w = (const float*)d_in[2];
    const float* conv_b = (const float*)d_in[3];
    const float* l1_w   = (const float*)d_in[4];
    const float* l1_b   = (const float*)d_in[5];
    const float* l2_w   = (const float*)d_in[6];
    const float* l2_b   = (const float*)d_in[7];

    float* out    = (float*)d_out;                               // (N,64,6,512)
    float* a4_out = out + (size_t)NB * COUTC * TT * VV;          // (N,1,512,512)

    char* ws = (char*)d_ws;
    __bf16* xc   = (__bf16*)ws;                                  // 12,582,912 B
    size_t off   = (size_t)NB * COUTC * TV * sizeof(__bf16);
    float* s_i   = (float*)(ws + off);   off += (size_t)NB * VV * sizeof(float);
    float* s_j   = (float*)(ws + off);   off += (size_t)NB * VV * sizeof(float);
    __bf16* a4T  = (__bf16*)(ws + off);                          // 16,777,216 B

    k1_conv_gemm <<<3072, 256, 0, stream>>>(x, conv_w, conv_b, xc);
    k2_scores    <<<(NB * VV) / 256, 256, 0, stream>>>(xc, l2_w, l2_b, l1_w, s_i, s_j);
    k3_softmax   <<<NB * VV, 256, 0, stream>>>(A, s_i, s_j, l1_b, a4_out);
    k3b_transpose<<<dim3(16, 16, NB), dim3(32, 8), 0, stream>>>(a4_out, a4T);
    k4_out_gemm  <<<768, 256, 0, stream>>>(xc, a4T, out);
}
